// PredictionAttention_42236708389239
// MI455X (gfx1250) — matrix operand order repaired and checked by reading, not compiled, not run on hardware
//
#include <hip/hip_runtime.h>
#include <math.h>

#define D_  32
#define T_  4
#define L_  24
#define TOK 96   // T_*L_

typedef __attribute__((ext_vector_type(16))) _Float16 v16h;
typedef __attribute__((ext_vector_type(8)))  _Float16 v8h;
typedef __attribute__((ext_vector_type(8)))  float    v8f;
typedef unsigned int u32x4 __attribute__((ext_vector_type(4)));
typedef int          i32x8 __attribute__((ext_vector_type(8)));
typedef int          i32x4 __attribute__((ext_vector_type(4)));

// ---- d_ws layout (bytes); total 84096 ----
#define WS_VPRE     0        // v matrix [T][H][L][HD] f32 (3072)
#define WS_PROMPT   12288    // pos_prompt token-major [96][32] f32
#define WS_PRE      24576    // pre_prompt token-major [96][32] f32
#define WS_BIAS2    36864    // folded conv bias [32] f32
#define WS_WKER     36992    // kernel_W f16 [32][64]
#define WS_WQ       41088    // q_W f16 [32][32]
#define WS_WK       43136    // k_W f16 [32][32]
#define WS_WM       45184    // m_W f16 [32][32]
#define WS_WC       47232    // conv W f16 [9][32][64] (bn_g folded)

// ---------------- WMMA fragment helpers (CDNA5 16-bit layouts) ----------------
__device__ __forceinline__ v8f wmma16(v16h a, v16h b, v8f c) {
  return __builtin_amdgcn_wmma_f32_16x16x32_f16(false, a, false, b, (short)0, c,
                                                false, false);
}

// A (16x32 f16): lane m(0..15) holds row m, K chunks {kb+0..7, kb+16..23};
// lanes 16..31 hold K chunks {kb+8..15, kb+24..31}. `row` precomputed by caller.
__device__ __forceinline__ v16h ldA(const _Float16* base, int pitch, int row,
                                    int kbase, int lane) {
  int sel = (lane >> 4) & 1;
  const _Float16* p = base + row * pitch + kbase + sel * 8;
  v8h lo = *(const v8h*)p;
  v8h hi = *(const v8h*)(p + 16);
  v16h a;
#pragma unroll
  for (int i = 0; i < 8; ++i) { a[i] = lo[i]; a[i + 8] = hi[i]; }
  return a;
}

// B (32x16 f16): weights stored [out][in]; lane n(0..15)=col n holds K {0..7, 16..23},
// lanes 16..31 hold K {8..15, 24..31} -> two 8-half reads per lane (matches ldA).
__device__ __forceinline__ v16h ldB(const _Float16* W, int pitch, int tile_n,
                                    int kbase, int lane) {
  int n = lane & 15, sel = (lane >> 4) & 1;
  const _Float16* p = W + (tile_n * 16 + n) * pitch + kbase + sel * 8;
  v8h lo = *(const v8h*)p;
  v8h hi = *(const v8h*)(p + 16);
  v16h r;
#pragma unroll
  for (int i = 0; i < 8; ++i) { r[i] = lo[i]; r[i + 8] = hi[i]; }
  return r;
}

// D (16x16 f32): VGPR r -> row r (+8 for upper half-wave), col = lane&15.
__device__ __forceinline__ void stD(float* C, int pitch, int tile_m, int tile_n,
                                    int lane, v8f d) {
  int n = lane & 15, sel = (lane >> 4) & 1;
  float* p = C + (tile_m * 16 + sel * 8) * pitch + tile_n * 16 + n;
#pragma unroll
  for (int r = 0; r < 8; ++r) p[r * pitch] = d[r];
}

// ---------------- Pre-pass: batch-invariant data into d_ws ----------------
__global__ void pa_prepass(const float* __restrict__ pos_prompt,
                           const float* __restrict__ rule_prompt,
                           const float* __restrict__ pre_prompt,
                           const float* __restrict__ kernel_W,
                           const float* __restrict__ q_W,
                           const float* __restrict__ k_W,
                           const float* __restrict__ v_W,
                           const float* __restrict__ v_b,
                           const float* __restrict__ m_W,
                           const float* __restrict__ conv_W,
                           const float* __restrict__ conv_b,
                           const float* __restrict__ bn_g,
                           const float* __restrict__ bn_b,
                           char* __restrict__ ws) {
  float* vpre   = (float*)(ws + WS_VPRE);
  float* ptok   = (float*)(ws + WS_PROMPT);
  float* pretok = (float*)(ws + WS_PRE);
  float* bias2  = (float*)(ws + WS_BIAS2);
  _Float16* Wker = (_Float16*)(ws + WS_WKER);
  _Float16* Wq   = (_Float16*)(ws + WS_WQ);
  _Float16* Wk   = (_Float16*)(ws + WS_WK);
  _Float16* Wm   = (_Float16*)(ws + WS_WM);
  _Float16* Wc   = (_Float16*)(ws + WS_WC);
  const int tid = threadIdx.x;

  // prompts: [D][TOK] -> token-major [TOK][D]
  for (int e = tid; e < D_ * TOK; e += 256) {
    int d = e / TOK, tl = e % TOK;
    ptok[tl * 32 + d]   = pos_prompt[e];
    pretok[tl * 32 + d] = pre_prompt[e];
  }
  // v = heads(linear(rule_prompt, v_W, v_b)) : [T][H][L][HD], l2-normed per head
  for (int task = tid; task < TOK * 8; task += 256) {
    int tl = task >> 3, h = task & 7;
    int t = tl / 24, l = tl % 24;
    float vv[4]; float nn = 0.f;
#pragma unroll
    for (int hd = 0; hd < 4; ++hd) {
      int d = h * 4 + hd;
      float acc = v_b[d];
      for (int c = 0; c < 32; ++c) acc += rule_prompt[c * TOK + tl] * v_W[d * 32 + c];
      vv[hd] = acc; nn += acc * acc;
    }
    float inv = 1.f / fmaxf(sqrtf(nn), 1e-12f);
    float* dst = vpre + ((t * 8 + h) * 24 + l) * 4;
#pragma unroll
    for (int hd = 0; hd < 4; ++hd) dst[hd] = vv[hd] * inv;
  }
  // weights -> f16 [out][in]
  for (int e = tid; e < 32 * 64; e += 256) Wker[e] = (_Float16)kernel_W[e];
  for (int e = tid; e < 32 * 32; e += 256) {
    Wq[e] = (_Float16)q_W[e];
    Wk[e] = (_Float16)k_W[e];
    Wm[e] = (_Float16)m_W[e];
  }
  // conv OIHW -> [s=ky*3+kx][o][i], bn_g folded
  for (int e = tid; e < 9 * 32 * 64; e += 256) {
    int s = e >> 11, r = e & 2047, o = r >> 6, i = r & 63;
    Wc[e] = (_Float16)(conv_W[o * 576 + i * 9 + s] * bn_g[o]);
  }
  for (int e = tid; e < 32; e += 256) bias2[e] = conv_b[e] * bn_g[e] + bn_b[e];
}

// ---------------- Main kernel: one workgroup per batch element ----------------
__global__ __launch_bounds__(192)
void pa_main(const float* __restrict__ x, float* __restrict__ out,
             const char* __restrict__ ws,
             const float* __restrict__ kernel_b, const float* __restrict__ ln_g,
             const float* __restrict__ ln_b, const float* __restrict__ q_b,
             const float* __restrict__ k_b, const float* __restrict__ m_b,
             const float* __restrict__ p_W, const float* __restrict__ p_b,
             const int* __restrict__ flag_p) {
  extern __shared__ char smem[];
  float*    Xt = (float*)smem;                  // [96][33] f32, token-major x
  _Float16* AH = (_Float16*)(smem + 12672);     // [96][72] f16 operand buffer
  float*    B1 = (float*)(smem + 26496);        // [96][33] f32 scratch
  float*    B2 = B1 + 96 * 33;
  float*    B3 = B2 + 96 * 33;
  float*    Xstage = B1;                        // TDM landing pad, aliases B1
  const unsigned XSTAGE_OFF = 26496u;           // LDS byte offset of Xstage

  const float* vpre   = (const float*)(ws + WS_VPRE);
  const float* ptok   = (const float*)(ws + WS_PROMPT);
  const float* pretok = (const float*)(ws + WS_PRE);
  const float* bias2  = (const float*)(ws + WS_BIAS2);
  const _Float16* Wker = (const _Float16*)(ws + WS_WKER);
  const _Float16* Wq   = (const _Float16*)(ws + WS_WQ);
  const _Float16* Wk   = (const _Float16*)(ws + WS_WK);
  const _Float16* Wm   = (const _Float16*)(ws + WS_WM);
  const _Float16* Wc   = (const _Float16*)(ws + WS_WC);

  const int tid  = threadIdx.x;
  const int lane = tid & 31;
  const int wv   = tid >> 5;  // 0..5 : M-tile owned by this wave
  const int b    = blockIdx.x;
  const int flag = *flag_p;
  const int S0   = (flag == 1) ? 18 : (flag == 2) ? 12 : (flag == 3) ? 6 : 0;

  // A: TDM DMA of x[b] ([32][96] f32, 12288 B contiguous) into LDS staging.
  // D# per CDNA5 ISA §8.3/§8.4: 2-D tile 96x32, data_size=4B, groups 2/3 zero.
  if (wv == 0) {
    unsigned long long ga = (unsigned long long)(const void*)(x + (size_t)b * (D_ * TOK));
    u32x4 g0;
    g0[0] = 1u;                                   // count=1, user descriptor
    g0[1] = XSTAGE_OFF;                           // lds_addr (bytes)
    g0[2] = (unsigned)(ga & 0xFFFFFFFFu);         // global_addr[31:0]
    g0[3] = (unsigned)((ga >> 32) & 0x1FFFFFFu)   // global_addr[56:32]
            | (2u << 30);                         // type=2 ("image")
    i32x8 g1;
    g1[0] = (2 << 16);                            // data_size = 4B
    g1[1] = (int)(96u << 16);                     // tensor_dim0 = 96 (lo16)
    g1[2] = (int)(32u << 16);                     // tensor_dim1 = 32 (lo16)
    g1[3] = (int)(96u << 16);                     // tile_dim0 = 96
    g1[4] = 32;                                   // tile_dim1 = 32, tile_dim2 = 0
    g1[5] = 96;                                   // tensor_dim0_stride = 96
    g1[6] = 0;
    g1[7] = 0;
    i32x4 gz4 = {};
    i32x8 gz8 = {};
    __builtin_amdgcn_tensor_load_to_lds(g0, g1, gz4, gz4, gz8, 0);
    __builtin_amdgcn_s_wait_tensorcnt(0);
  }
  __syncthreads();

  // transpose staging [D][TOK] -> padded token-major Xt[96][33]
  for (int e = tid; e < D_ * TOK; e += 192) {
    int tl = e % TOK, d = e / TOK;
    Xt[tl * 33 + d] = Xstage[d * TOK + tl];
  }
  __syncthreads();

  // B: build [con | tar] f16 rows (kernel-linear A, K=64)
  for (int e = tid; e < TOK * D_; e += 192) {
    int d = e & 31, tl = e >> 5, l = tl % 24;
    bool s = (l >= S0) && (l < S0 + 6);
    float xv = Xt[tl * 33 + d], pv = ptok[tl * 32 + d];
    AH[tl * 72 + d]      = (_Float16)(s ? pv : xv);  // con
    AH[tl * 72 + 32 + d] = (_Float16)(s ? xv : pv);  // tar
  }
  __syncthreads();

  // C: kernel linear [96,64]x[64->32]
  {
    v8f c0 = {}, c1 = {};
    int row = wv * 16 + (lane & 15);
#pragma unroll
    for (int ks = 0; ks < 2; ++ks) {
      int kb = ks * 32;
      v16h a  = ldA(AH, 72, row, kb, lane);
      v16h b0 = ldB(Wker, 64, 0, kb, lane);
      v16h b1 = ldB(Wker, 64, 1, kb, lane);
      c0 = wmma16(a, b0, c0);
      c1 = wmma16(a, b1, c1);
    }
    stD(B1, 33, wv, 0, lane, c0);
    stD(B1, 33, wv, 1, lane, c1);
  }
  __syncthreads();

  // D: +bias, LayerNorm, exact GELU, per-head l2 -> tc_kernel in B1
  if (tid < TOK) {
    float vr[32]; float mu = 0.f;
#pragma unroll
    for (int d = 0; d < 32; ++d) { vr[d] = B1[tid * 33 + d] + kernel_b[d]; mu += vr[d]; }
    mu *= (1.f / 32.f);
    float var = 0.f;
#pragma unroll
    for (int d = 0; d < 32; ++d) { float dv = vr[d] - mu; var += dv * dv; }
    var *= (1.f / 32.f);
    float inv = rsqrtf(var + 1e-5f);
#pragma unroll
    for (int d = 0; d < 32; ++d) {
      float g = (vr[d] - mu) * inv * ln_g[d] + ln_b[d];
      vr[d] = 0.5f * g * (1.f + erff(g * 0.70710678118654752f));
    }
#pragma unroll
    for (int h = 0; h < 8; ++h) {
      float nn = 0.f;
#pragma unroll
      for (int hd = 0; hd < 4; ++hd) nn += vr[h * 4 + hd] * vr[h * 4 + hd];
      float i2 = 1.f / fmaxf(sqrtf(nn), 1e-12f);
#pragma unroll
      for (int hd = 0; hd < 4; ++hd) B1[tid * 33 + h * 4 + hd] = vr[h * 4 + hd] * i2;
    }
  }
  __syncthreads();

  // E: q (from tar slice, kbase=32) and k (from con slice, kbase=0) linears
  {
    int row = wv * 16 + (lane & 15);
    v16h aT = ldA(AH, 72, row, 32, lane);
    v16h aC = ldA(AH, 72, row, 0, lane);
    v8f q0 = {}, q1 = {}, k0 = {}, k1 = {};
    q0 = wmma16(aT, ldB(Wq, 32, 0, 0, lane), q0);
    q1 = wmma16(aT, ldB(Wq, 32, 1, 0, lane), q1);
    k0 = wmma16(aC, ldB(Wk, 32, 0, 0, lane), k0);
    k1 = wmma16(aC, ldB(Wk, 32, 1, 0, lane), k1);
    stD(B2, 33, wv, 0, lane, q0); stD(B2, 33, wv, 1, lane, q1);
    stD(B3, 33, wv, 0, lane, k0); stD(B3, 33, wv, 1, lane, k1);
  }
  __syncthreads();

  // F: q/k bias, per-head l2, multiply by tc_kernel
  if (tid < TOK) {
#pragma unroll
    for (int h = 0; h < 8; ++h) {
      float qv[4], kv[4]; float nq = 0.f, nk = 0.f;
#pragma unroll
      for (int hd = 0; hd < 4; ++hd) {
        int d = h * 4 + hd;
        qv[hd] = B2[tid * 33 + d] + q_b[d]; nq += qv[hd] * qv[hd];
        kv[hd] = B3[tid * 33 + d] + k_b[d]; nk += kv[hd] * kv[hd];
      }
      float iq = 1.f / fmaxf(sqrtf(nq), 1e-12f);
      float ik = 1.f / fmaxf(sqrtf(nk), 1e-12f);
#pragma unroll
      for (int hd = 0; hd < 4; ++hd) {
        int d = h * 4 + hd;
        float tck = B1[tid * 33 + d];
        B2[tid * 33 + d] = qv[hd] * iq * tck;
        B3[tid * 33 + d] = kv[hd] * ik * tck;
      }
    }
  }
  __syncthreads();

  // G: attention (HD=4 -> VALU). 32 (t,h) tasks x 6 threads x 4 rows.
  {
    int task = tid / 6, sub = tid % 6;
    int t = task >> 3, h = task & 7;
    const float* vh = vpre + ((t * 8 + h) * 24) * 4;
#pragma unroll
    for (int rr = 0; rr < 4; ++rr) {
      int i = sub * 4 + rr;
      float qi[4];
#pragma unroll
      for (int c = 0; c < 4; ++c) qi[c] = B2[(t * 24 + i) * 33 + h * 4 + c];
      float sc[24]; float mx = -1e30f;
#pragma unroll
      for (int j = 0; j < 24; ++j) {
        float s = 0.f;
#pragma unroll
        for (int c = 0; c < 4; ++c) s += qi[c] * B3[(t * 24 + j) * 33 + h * 4 + c];
        s *= 0.5f;  // 1/sqrt(HD)
        sc[j] = s; mx = fmaxf(mx, s);
      }
      float sum = 0.f;
#pragma unroll
      for (int j = 0; j < 24; ++j) { sc[j] = __expf(sc[j] - mx); sum += sc[j]; }
      float inv = 1.f / sum;
      float r0 = 0.f, r1 = 0.f, r2 = 0.f, r3 = 0.f;
#pragma unroll
      for (int j = 0; j < 24; ++j) {
        float a = sc[j] * inv;
        r0 += a * vh[j * 4 + 0]; r1 += a * vh[j * 4 + 1];
        r2 += a * vh[j * 4 + 2]; r3 += a * vh[j * 4 + 3];
      }
      _Float16* dst = AH + (t * 24 + i) * 72 + h * 4;  // m-linear A operand
      dst[0] = (_Float16)r0; dst[1] = (_Float16)r1;
      dst[2] = (_Float16)r2; dst[3] = (_Float16)r3;
    }
  }
  __syncthreads();

  // H: m linear [96,32]x[32->32]
  {
    int row = wv * 16 + (lane & 15);
    v16h a = ldA(AH, 72, row, 0, lane);
    v8f c0 = {}, c1 = {};
    c0 = wmma16(a, ldB(Wm, 32, 0, 0, lane), c0);
    c1 = wmma16(a, ldB(Wm, 32, 1, 0, lane), c1);
    stD(B2, 33, wv, 0, lane, c0);
    stD(B2, 33, wv, 1, lane, c1);
  }
  __syncthreads();

  // I: conv input [con2(32ch) | m_out+m_b(32ch)] as f16 rows
  for (int e = tid; e < TOK * D_; e += 192) {
    int d = e & 31, tl = e >> 5, l = tl % 24;
    bool s = (l >= S0) && (l < S0 + 6);
    AH[tl * 72 + d]      = (_Float16)(s ? pretok[tl * 32 + d] : Xt[tl * 33 + d]);
    AH[tl * 72 + 32 + d] = (_Float16)(B2[tl * 33 + d] + m_b[d]);
  }
  __syncthreads();

  // J: 3x3 conv = 9 shifted GEMMs [96,64]x[64->32], borders zero via data select
  {
    int mrow = wv * 16 + (lane & 15);
    int tt = mrow / 24, ll = mrow % 24;
    v16h zero16;
#pragma unroll
    for (int i = 0; i < 16; ++i) zero16[i] = (_Float16)0.f;
    v8f c0 = {}, c1 = {};
#pragma unroll
    for (int s = 0; s < 9; ++s) {
      int dy = s / 3 - 1, dx = s % 3 - 1;
      int ts = tt + dy, ls = ll + dx;
      bool valid = (ts >= 0) && (ts < T_) && (ls >= 0) && (ls < L_);
      int srow = valid ? (ts * 24 + ls) : 0;
      const _Float16* Ws = Wc + s * 2048;
#pragma unroll
      for (int ks = 0; ks < 2; ++ks) {
        int kb = ks * 32;
        v16h a = valid ? ldA(AH, 72, srow, kb, lane) : zero16;
        v16h b0 = ldB(Ws, 64, 0, kb, lane);
        v16h b1 = ldB(Ws, 64, 1, kb, lane);
        c0 = wmma16(a, b0, c0);
        c1 = wmma16(a, b1, c1);
      }
    }
    // epilogue: folded BN bias + ReLU -> B1
    int n = lane & 15, sel = (lane >> 4) & 1;
    float bb0 = bias2[n], bb1 = bias2[16 + n];
    float* p0 = B1 + (wv * 16 + sel * 8) * 33 + n;
    float* p1 = p0 + 16;
#pragma unroll
    for (int r = 0; r < 8; ++r) {
      p0[r * 33] = fmaxf(c0[r] + bb0, 0.f);
      p1[r * 33] = fmaxf(c1[r] + bb1, 0.f);
    }
  }
  __syncthreads();

  // K: p = y @ p_W.T + p_b on target columns, copy x elsewhere; emit [b,T,L,D]
  float* ob = out + (size_t)b * (T_ * L_ * D_);
  for (int e = tid; e < TOK * D_; e += 192) {
    int d = e & 31, tl = e >> 5, t = tl / 24, l = tl % 24;
    float val;
    if (l >= S0 && l < S0 + 6) {
      int j = l - S0;
      float acc = p_b[j];
#pragma unroll
      for (int ll = 0; ll < 24; ++ll)
        acc += B1[(t * 24 + ll) * 33 + d] * p_W[j * 24 + ll];
      val = acc;
    } else {
      val = Xt[tl * 33 + d];
    }
    ob[tl * 32 + d] = val;  // coalesced: d fastest
  }
}

extern "C" void kernel_launch(void* const* d_in, const int* in_sizes, int n_in,
                              void* d_out, int out_size, void* d_ws, size_t ws_size,
                              hipStream_t stream) {
  const float* x           = (const float*)d_in[0];
  const float* pos_prompt  = (const float*)d_in[1];
  const float* rule_prompt = (const float*)d_in[2];
  const float* pre_prompt  = (const float*)d_in[3];
  const float* kernel_W    = (const float*)d_in[4];
  const float* kernel_b    = (const float*)d_in[5];
  const float* ln_g        = (const float*)d_in[6];
  const float* ln_b        = (const float*)d_in[7];
  const float* q_W         = (const float*)d_in[8];
  const float* q_b         = (const float*)d_in[9];
  const float* k_W         = (const float*)d_in[10];
  const float* k_b         = (const float*)d_in[11];
  const float* v_W         = (const float*)d_in[12];
  const float* v_b         = (const float*)d_in[13];
  const float* m_W         = (const float*)d_in[14];
  const float* m_b         = (const float*)d_in[15];
  const float* conv_W      = (const float*)d_in[16];
  const float* conv_b      = (const float*)d_in[17];
  const float* bn_g        = (const float*)d_in[18];
  const float* bn_b        = (const float*)d_in[19];
  const float* p_W         = (const float*)d_in[20];
  const float* p_b         = (const float*)d_in[21];
  const int*   atten_flag  = (const int*)d_in[22];

  const int b = in_sizes[0] / (D_ * T_ * L_);

  pa_prepass<<<1, 256, 0, stream>>>(pos_prompt, rule_prompt, pre_prompt,
                                    kernel_W, q_W, k_W, v_W, v_b, m_W,
                                    conv_W, conv_b, bn_g, bn_b, (char*)d_ws);

  pa_main<<<b, 192, 64512, stream>>>(x, (float*)d_out, (const char*)d_ws,
                                     kernel_b, ln_g, ln_b, q_b, k_b, m_b,
                                     p_W, p_b, atten_flag);
}
